// CrossAttention_70574902608281
// MI455X (gfx1250) — matrix operand order repaired and checked by reading, not compiled, not run on hardware
//
#include <hip/hip_runtime.h>

#define E_DIM 512
#define NH 8
#define HD 64
#define LQ 1024
#define LKV 4096
#define NB 4
#define LN_EPS 1e-5f

typedef __attribute__((ext_vector_type(16))) __bf16 v16bf;
typedef __attribute__((ext_vector_type(2))) __bf16 v2bf;
typedef __attribute__((ext_vector_type(8))) float v8f;
typedef __attribute__((ext_vector_type(8))) unsigned int v8u;
typedef __attribute__((ext_vector_type(4))) int v4i;

#if __has_builtin(__builtin_amdgcn_global_load_async_to_lds_b128) && \
    __has_builtin(__builtin_amdgcn_s_wait_asynccnt)
#define HAVE_ASYNC_LDS 1
#endif

static __device__ __forceinline__ unsigned short f2bf(float f) {
  unsigned int u = __builtin_bit_cast(unsigned int, f);
  u += 0x7fffu + ((u >> 16) & 1u);  // round-to-nearest-even
  return (unsigned short)(u >> 16);
}

// packed f32x2 -> bf16x2 (one v_cvt_pk_bf16_f32 when available)
static __device__ __forceinline__ unsigned int f2bf2(float lo, float hi) {
#if __has_builtin(__builtin_amdgcn_cvt_pk_bf16_f32)
  v2bf r = __builtin_amdgcn_cvt_pk_bf16_f32(lo, hi);
  return __builtin_bit_cast(unsigned int, r);
#else
  return (unsigned int)f2bf(lo) | ((unsigned int)f2bf(hi) << 16);
#endif
}

#ifdef HAVE_ASYNC_LDS
static __device__ __forceinline__ void async_copy16(const void* g, void* l) {
  __builtin_amdgcn_global_load_async_to_lds_b128(
      (__attribute__((address_space(1))) v4i*)g,
      (__attribute__((address_space(3))) v4i*)l, 0, 0);
}
#endif

// A-fragment (16x32 bf16) for V_WMMA_F32_16X16X32_BF16.
// lane: M = lane&15 ; VGPR j holds K = (j>=4?16:0) + (lane>=16?8:0) + (j&3)*2 + {0,1}
static __device__ __forceinline__ v16bf load_afrag(const unsigned short* base, int ldh) {
  const int lane = threadIdx.x & 31;
  const int m = lane & 15, hi = lane >> 4;
  v8u u;
#pragma unroll
  for (int j = 0; j < 8; ++j) {
    const int kk = ((j & 4) << 2) + hi * 8 + ((j & 3) << 1);
    u[j] = *(const unsigned int*)(base + (size_t)m * ldh + kk);
  }
  return __builtin_bit_cast(v16bf, u);
}

// B-fragment (32x16 bf16): frag column N = lane&15 maps to tile row r of base[r][k];
// VGPR j holds K = (j>=4?16:0) + (lane>=16?8:0) + (j&3)*2 + {0,1}
static __device__ __forceinline__ v16bf load_bfrag(const unsigned short* base, int ldh) {
  const int lane = threadIdx.x & 31;
  const int n = lane & 15, hi = lane >> 4;
  v8u u;
#pragma unroll
  for (int j = 0; j < 8; ++j)
    u[j] = *(const unsigned int*)(base + (size_t)n * ldh + ((j & 4) << 2) + hi * 8 + ((j & 3) << 1));
  return __builtin_bit_cast(v16bf, u);
}

static __device__ __forceinline__ v8f wmma_bf16(v16bf a, v16bf b, v8f c) {
  return __builtin_amdgcn_wmma_f32_16x16x32_bf16(false, a, false, b, (short)0, c,
                                                 false, false);
}

// ---------------------------------------------------------------------------
// Kernel 1: C_bf16[M,512] = A_f32[M,512] @ W_f32[512,512]^T + bias
// block 256 thr (8 waves): 64x64 tile; wave = 16(M) x 32(N); K-step 32.
// ---------------------------------------------------------------------------
__global__ __launch_bounds__(256) void k_proj_gemm(const float* __restrict__ A,
                                                   const float* __restrict__ W,
                                                   const float* __restrict__ bias,
                                                   unsigned short* __restrict__ C) {
  __shared__ __align__(16) unsigned short As[64][32];
  __shared__ __align__(16) unsigned short Ws[64][32];
  const int tid = threadIdx.x;
  const int wave = tid >> 5, lane = tid & 31;
  const int wm = wave & 3, wn = wave >> 2;
  const int m0 = blockIdx.y * 64, n0 = blockIdx.x * 64;
  v8f c0 = {}, c1 = {};
  for (int k0 = 0; k0 < E_DIM; k0 += 32) {
#pragma unroll
    for (int i = 0; i < 4; ++i) {  // 1024 bf16-pairs per tile
      const int e2 = tid + i * 256;
      const int r = e2 >> 4, kk = (e2 & 15) * 2;
      const float2 av = *(const float2*)&A[(size_t)(m0 + r) * E_DIM + k0 + kk];
      const float2 wv = *(const float2*)&W[(size_t)(n0 + r) * E_DIM + k0 + kk];
      *(unsigned int*)&As[r][kk] = f2bf2(av.x, av.y);
      *(unsigned int*)&Ws[r][kk] = f2bf2(wv.x, wv.y);
    }
    __syncthreads();
    v16bf a = load_afrag(&As[wm * 16][0], 32);
    v16bf b0 = load_bfrag(&Ws[wn * 32][0], 32);
    v16bf b1 = load_bfrag(&Ws[wn * 32 + 16][0], 32);
    c0 = wmma_bf16(a, b0, c0);
    c1 = wmma_bf16(a, b1, c1);
    __syncthreads();
  }
  const int nlo = lane & 15, hi = lane >> 4;
  const float bb0 = bias[n0 + wn * 32 + nlo];
  const float bb1 = bias[n0 + wn * 32 + 16 + nlo];
#pragma unroll
  for (int vr = 0; vr < 8; ++vr) {
    const int mg = m0 + wm * 16 + vr + hi * 8;
    C[(size_t)mg * E_DIM + n0 + wn * 32 + nlo] = f2bf(c0[vr] + bb0);
    C[(size_t)mg * E_DIM + n0 + wn * 32 + 16 + nlo] = f2bf(c1[vr] + bb1);
  }
}

// ---------------------------------------------------------------------------
// Kernel 2: softmax stats. grid (LQ/64, B*H); block 128 (4 waves x 16 q rows).
// Streams S = Q K^T / sqrt(D), online max / sum-exp per row -> (m, 1/l).
// ---------------------------------------------------------------------------
__global__ __launch_bounds__(128) void k_attn_stats(const unsigned short* __restrict__ Qb,
                                                    const unsigned short* __restrict__ Kb,
                                                    float* __restrict__ Mst,
                                                    float* __restrict__ Lst) {
  __shared__ __align__(16) unsigned short Ks[16][64];
  const int tid = threadIdx.x;
  const int wave = tid >> 5, lane = tid & 31, hi = lane >> 4;
  const int bh = blockIdx.y, b = bh >> 3, h = bh & 7;
  const int q0 = blockIdx.x * 64 + wave * 16;
  const unsigned short* qbase = Qb + ((size_t)(b * LQ + q0) * E_DIM + h * HD);
  const v16bf aq0 = load_afrag(qbase, E_DIM);
  const v16bf aq1 = load_afrag(qbase + 32, E_DIM);
  float mi[8], li[8];
#pragma unroll
  for (int vr = 0; vr < 8; ++vr) { mi[vr] = -1e30f; li[vr] = 0.f; }
  const int sr = tid >> 3, sc8 = tid & 7;  // staging coords: 128 x b128
  for (int kv0 = 0; kv0 < LKV; kv0 += 16) {
    const unsigned short* gsrc =
        Kb + ((size_t)(b * LKV + kv0 + sr) * E_DIM + h * HD) + sc8 * 8;
#ifdef HAVE_ASYNC_LDS
    async_copy16(gsrc, &Ks[sr][sc8 * 8]);
    __builtin_amdgcn_s_wait_asynccnt(0);
#else
    *(uint4*)&Ks[sr][sc8 * 8] = *(const uint4*)gsrc;
#endif
    __syncthreads();
    v16bf bk0 = load_bfrag(&Ks[0][0], 64);
    v16bf bk1 = load_bfrag(&Ks[0][32], 64);
    v8f z = {};
    v8f s = wmma_bf16(aq0, bk0, z);
    s = wmma_bf16(aq1, bk1, s);
#pragma unroll
    for (int vr = 0; vr < 8; ++vr) {
      const float sv = s[vr] * 0.125f;
      float mx = sv;
#pragma unroll
      for (int off = 1; off < 16; off <<= 1) mx = fmaxf(mx, __shfl_xor(mx, off, 32));
      const float mnew = fmaxf(mi[vr], mx);
      float rs = __expf(sv - mnew);
#pragma unroll
      for (int off = 1; off < 16; off <<= 1) rs += __shfl_xor(rs, off, 32);
      li[vr] = li[vr] * __expf(mi[vr] - mnew) + rs;
      mi[vr] = mnew;
    }
    __syncthreads();
  }
  if ((lane & 15) == 0) {
#pragma unroll
    for (int vr = 0; vr < 8; ++vr) {
      const size_t idx = (size_t)bh * LQ + q0 + vr + hi * 8;
      Mst[idx] = mi[vr];
      Lst[idx] = 1.0f / li[vr];
    }
  }
}

// ---------------------------------------------------------------------------
// Kernel 3: head-averaged attention weights -> d_out (no atomics).
// grid (LQ/64, B); block 128. One block owns 64 q rows; loops all 8 heads.
// This kernel writes the 64MB output that dominates the roofline.
// ---------------------------------------------------------------------------
__global__ __launch_bounds__(128) void k_attn_weights(const unsigned short* __restrict__ Qb,
                                                      const unsigned short* __restrict__ Kb,
                                                      const float* __restrict__ Mst,
                                                      const float* __restrict__ Lst,
                                                      float* __restrict__ Wout) {
  __shared__ __align__(16) unsigned short Ksa[16][E_DIM];  // 16 kv rows, all heads
  const int tid = threadIdx.x;
  const int wave = tid >> 5, lane = tid & 31;
  const int nlo = lane & 15, hi = lane >> 4;
  const int b = blockIdx.y;
  const int q0 = blockIdx.x * 64 + wave * 16;
  v16bf aq[NH][2];
  float ms[NH][8], ls[NH][8];
#pragma unroll
  for (int h = 0; h < NH; ++h) {
    const unsigned short* qbase = Qb + ((size_t)(b * LQ + q0) * E_DIM + h * HD);
    aq[h][0] = load_afrag(qbase, E_DIM);
    aq[h][1] = load_afrag(qbase + 32, E_DIM);
#pragma unroll
    for (int vr = 0; vr < 8; ++vr) {
      const size_t idx = ((size_t)(b * NH + h)) * LQ + q0 + vr + hi * 8;
      ms[h][vr] = Mst[idx];
      ls[h][vr] = Lst[idx];
    }
  }
  for (int kv0 = 0; kv0 < LKV; kv0 += 16) {
    const unsigned short* slab = Kb + (size_t)(b * LKV + kv0) * E_DIM;
    {  // stage 16x512 slab: 1024 x b128, 8 per thread
#pragma unroll
      for (int i = 0; i < 8; ++i) {
        const int e = tid + i * 128;
        const int r = e >> 6, c8 = e & 63;
        const unsigned short* gsrc = slab + (size_t)r * E_DIM + c8 * 8;
#ifdef HAVE_ASYNC_LDS
        async_copy16(gsrc, &Ksa[r][c8 * 8]);
#else
        *(uint4*)&Ksa[r][c8 * 8] = *(const uint4*)gsrc;
#endif
      }
#ifdef HAVE_ASYNC_LDS
      __builtin_amdgcn_s_wait_asynccnt(0);
#endif
    }
    if (kv0 + 16 < LKV)  // prefetch next slab (16KB) into the cache hierarchy
      __builtin_prefetch(slab + (size_t)16 * E_DIM + (size_t)tid * 64, 0, 1);
    __syncthreads();
    v8f wsum = {};
#pragma unroll
    for (int h = 0; h < NH; ++h) {
      v16bf bk0 = load_bfrag(&Ksa[0][h * HD], E_DIM);
      v16bf bk1 = load_bfrag(&Ksa[0][h * HD + 32], E_DIM);
      v8f z = {};
      v8f s = wmma_bf16(aq[h][0], bk0, z);
      s = wmma_bf16(aq[h][1], bk1, s);
#pragma unroll
      for (int vr = 0; vr < 8; ++vr)
        wsum[vr] += __expf(s[vr] * 0.125f - ms[h][vr]) * ls[h][vr];
    }
#pragma unroll
    for (int vr = 0; vr < 8; ++vr) {
      const int r = q0 + vr + hi * 8;
      Wout[((size_t)b * LQ + r) * LKV + kv0 + nlo] = wsum[vr] * 0.125f;  // /H
    }
    __syncthreads();
  }
}

// ---------------------------------------------------------------------------
// Kernel 4: ctx = softmax(S) @ V  (flash-style, stats precomputed).
// grid (LQ/64, B*H); block 128. kv-step 32; P goes through wave-private LDS
// to convert C-frag layout -> A-frag layout; V staged transposed.
// ---------------------------------------------------------------------------
__global__ __launch_bounds__(128) void k_attn_ctx(const unsigned short* __restrict__ Qb,
                                                  const unsigned short* __restrict__ Kb,
                                                  const unsigned short* __restrict__ Vb,
                                                  const float* __restrict__ Mst,
                                                  const float* __restrict__ Lst,
                                                  unsigned short* __restrict__ Ctxb) {
  __shared__ __align__(16) unsigned short Ks[32][64];
  __shared__ __align__(16) unsigned short Vts[64][32];  // transposed [d][kv]
  __shared__ __align__(16) unsigned short Ps[4][16][32];
  const int tid = threadIdx.x;
  const int wave = tid >> 5, lane = tid & 31;
  const int nlo = lane & 15, hi = lane >> 4;
  const int bh = blockIdx.y, b = bh >> 3, h = bh & 7;
  const int q0 = blockIdx.x * 64 + wave * 16;
  const unsigned short* qbase = Qb + ((size_t)(b * LQ + q0) * E_DIM + h * HD);
  const v16bf aq0 = load_afrag(qbase, E_DIM);
  const v16bf aq1 = load_afrag(qbase + 32, E_DIM);
  float ms[8], ls[8];
#pragma unroll
  for (int vr = 0; vr < 8; ++vr) {
    const size_t idx = (size_t)bh * LQ + q0 + vr + hi * 8;
    ms[vr] = Mst[idx];
    ls[vr] = Lst[idx];
  }
  v8f acc[4] = {};
  for (int kv0 = 0; kv0 < LKV; kv0 += 32) {
    {  // K tile 32x64: 256 x b128 -> 2 per thread
#pragma unroll
      for (int i = 0; i < 2; ++i) {
        const int e = tid + i * 128;
        const int r = e >> 3, c8 = e & 7;
        const unsigned short* gsrc =
            Kb + ((size_t)(b * LKV + kv0 + r) * E_DIM + h * HD) + c8 * 8;
#ifdef HAVE_ASYNC_LDS
        async_copy16(gsrc, &Ks[r][c8 * 8]);
#else
        *(uint4*)&Ks[r][c8 * 8] = *(const uint4*)gsrc;
#endif
      }
#ifdef HAVE_ASYNC_LDS
      __builtin_amdgcn_s_wait_asynccnt(0);
#endif
      // V tile transposed, scalar scatter
#pragma unroll
      for (int i = 0; i < 16; ++i) {
        const int e = tid + i * 128;
        const int r = e >> 6, c = e & 63;
        Vts[c][r] = Vb[((size_t)(b * LKV + kv0 + r) * E_DIM + h * HD) + c];
      }
    }
    if (kv0 + 32 < LKV)
      __builtin_prefetch(Kb + ((size_t)(b * LKV + kv0 + 32 + (tid >> 2)) * E_DIM + h * HD),
                         0, 1);
    __syncthreads();
#pragma unroll
    for (int sub = 0; sub < 2; ++sub) {
      v16bf bk0 = load_bfrag(&Ks[sub * 16][0], 64);
      v16bf bk1 = load_bfrag(&Ks[sub * 16][32], 64);
      v8f z = {};
      v8f s = wmma_bf16(aq0, bk0, z);
      s = wmma_bf16(aq1, bk1, s);
#pragma unroll
      for (int vr = 0; vr < 8; ++vr) {
        const float p = __expf(s[vr] * 0.125f - ms[vr]) * ls[vr];
        Ps[wave][vr + hi * 8][sub * 16 + nlo] = f2bf(p);  // LDS in-order per wave
      }
    }
    v16bf ap = load_afrag(&Ps[wave][0][0], 32);
#pragma unroll
    for (int dt = 0; dt < 4; ++dt) {
      v16bf bv = load_bfrag(&Vts[dt * 16][0], 32);
      acc[dt] = wmma_bf16(ap, bv, acc[dt]);
    }
    __syncthreads();
  }
#pragma unroll
  for (int dt = 0; dt < 4; ++dt)
#pragma unroll
    for (int vr = 0; vr < 8; ++vr) {
      const int r = q0 + vr + hi * 8;
      Ctxb[((size_t)(b * LQ + r)) * E_DIM + h * HD + dt * 16 + nlo] = f2bf(acc[dt][vr]);
    }
}

// ---------------------------------------------------------------------------
// Kernel 5: out = LayerNorm(query + ctx @ Wo^T + bo) * gamma + beta.
// grid (B*LQ/16); block 256 (8 waves x 64 cols). Fused residual + LN.
// ---------------------------------------------------------------------------
__global__ __launch_bounds__(256) void k_oproj_ln(const unsigned short* __restrict__ Ctxb,
                                                  const float* __restrict__ Wo,
                                                  const float* __restrict__ bo,
                                                  const float* __restrict__ query,
                                                  const float* __restrict__ gamma,
                                                  const float* __restrict__ beta,
                                                  float* __restrict__ Out) {
  __shared__ __align__(16) unsigned short Ax[16][32];
  __shared__ __align__(16) unsigned short Ws[E_DIM][32];
  __shared__ float redS[16][8];
  __shared__ float redQ[16][8];
  __shared__ float muS[16];
  __shared__ float rsS[16];
  const int tid = threadIdx.x;
  const int wave = tid >> 5, lane = tid & 31;
  const int nlo = lane & 15, hi = lane >> 4;
  const int R0 = blockIdx.x * 16;
  v8f c[4] = {};
  for (int k0 = 0; k0 < E_DIM; k0 += 32) {
    {  // stage 16x32 ctx tile (1 b32 per thread)
      const int r = tid >> 4, kk = (tid & 15) * 2;
      *(unsigned int*)&Ax[r][kk] =
          *(const unsigned int*)(Ctxb + ((size_t)(R0 + r) * E_DIM + k0 + kk));
    }
#pragma unroll 4
    for (int i = 0; i < 32; ++i) {  // stage 512x32 Wo slice as bf16 pairs
      const int e2 = tid + i * 256;
      const int n = e2 >> 4, kk = (e2 & 15) * 2;
      const float2 wv = *(const float2*)&Wo[(size_t)n * E_DIM + k0 + kk];
      *(unsigned int*)&Ws[n][kk] = f2bf2(wv.x, wv.y);
    }
    __syncthreads();
    v16bf a = load_afrag(&Ax[0][0], 32);
#pragma unroll
    for (int dt = 0; dt < 4; ++dt) {
      v16bf bw = load_bfrag(&Ws[wave * 64 + dt * 16][0], 32);
      c[dt] = wmma_bf16(a, bw, c[dt]);
    }
    __syncthreads();
  }
  float x[4][8];
#pragma unroll
  for (int dt = 0; dt < 4; ++dt) {
    const int ng = wave * 64 + dt * 16 + nlo;
    const float bb = bo[ng];
#pragma unroll
    for (int vr = 0; vr < 8; ++vr) {
      const int r = R0 + vr + hi * 8;
      x[dt][vr] = c[dt][vr] + bb + query[(size_t)r * E_DIM + ng];
    }
  }
#pragma unroll
  for (int vr = 0; vr < 8; ++vr) {
    float s1 = x[0][vr] + x[1][vr] + x[2][vr] + x[3][vr];
    float s2 = x[0][vr] * x[0][vr] + x[1][vr] * x[1][vr] + x[2][vr] * x[2][vr] +
               x[3][vr] * x[3][vr];
#pragma unroll
    for (int off = 1; off < 16; off <<= 1) {
      s1 += __shfl_xor(s1, off, 32);
      s2 += __shfl_xor(s2, off, 32);
    }
    if (nlo == 0) {
      redS[vr + hi * 8][wave] = s1;
      redQ[vr + hi * 8][wave] = s2;
    }
  }
  __syncthreads();
  if (tid < 16) {
    float s1 = 0.f, s2 = 0.f;
#pragma unroll
    for (int w = 0; w < 8; ++w) { s1 += redS[tid][w]; s2 += redQ[tid][w]; }
    const float mu = s1 * (1.0f / E_DIM);
    const float var = s2 * (1.0f / E_DIM) - mu * mu;
    muS[tid] = mu;
    rsS[tid] = rsqrtf(var + LN_EPS);
  }
  __syncthreads();
#pragma unroll
  for (int dt = 0; dt < 4; ++dt) {
    const int ng = wave * 64 + dt * 16 + nlo;
    const float g = gamma[ng], bt = beta[ng];
#pragma unroll
    for (int vr = 0; vr < 8; ++vr) {
      const int rr = vr + hi * 8;
      Out[(size_t)(R0 + rr) * E_DIM + ng] = (x[dt][vr] - muS[rr]) * rsS[rr] * g + bt;
    }
  }
}

// ---------------------------------------------------------------------------
extern "C" void kernel_launch(void* const* d_in, const int* in_sizes, int n_in,
                              void* d_out, int out_size, void* d_ws, size_t ws_size,
                              hipStream_t stream) {
  const float* query = (const float*)d_in[0];
  const float* key_value = (const float*)d_in[1];
  const float* w_q = (const float*)d_in[2];
  const float* w_k = (const float*)d_in[3];
  const float* w_v = (const float*)d_in[4];
  const float* b_q = (const float*)d_in[5];
  const float* b_k = (const float*)d_in[6];
  const float* b_v = (const float*)d_in[7];
  const float* w_o = (const float*)d_in[8];
  const float* b_o = (const float*)d_in[9];
  const float* ln_g = (const float*)d_in[10];
  const float* ln_b = (const float*)d_in[11];

  char* ws = (char*)d_ws;
  // workspace layout (bytes)
  unsigned short* Qb = (unsigned short*)(ws + 0);            //  4,194,304
  unsigned short* Kb = (unsigned short*)(ws + 4194304);      // 16,777,216
  unsigned short* Vb = (unsigned short*)(ws + 20971520);     // 16,777,216
  float* Mst = (float*)(ws + 37748736);                      //    131,072
  float* Lst = (float*)(ws + 37879808);                      //    131,072
  unsigned short* Ctxb = (unsigned short*)(ws + 38010880);   //  4,194,304

  float* out = (float*)d_out;                    // [B,LQ,E]
  float* attn = out + (size_t)NB * LQ * E_DIM;   // [B,LQ,LKV]

  k_proj_gemm<<<dim3(E_DIM / 64, (NB * LQ) / 64), 256, 0, stream>>>(query, w_q, b_q, Qb);
  k_proj_gemm<<<dim3(E_DIM / 64, (NB * LKV) / 64), 256, 0, stream>>>(key_value, w_k, b_k, Kb);
  k_proj_gemm<<<dim3(E_DIM / 64, (NB * LKV) / 64), 256, 0, stream>>>(key_value, w_v, b_v, Vb);
  k_attn_stats<<<dim3(LQ / 64, NB * NH), 128, 0, stream>>>(Qb, Kb, Mst, Lst);
  k_attn_weights<<<dim3(LQ / 64, NB), 128, 0, stream>>>(Qb, Kb, Mst, Lst, attn);
  k_attn_ctx<<<dim3(LQ / 64, NB * NH), 128, 0, stream>>>(Qb, Kb, Vb, Mst, Lst, Ctxb);
  k_oproj_ln<<<dim3((NB * LQ) / 16), 256, 0, stream>>>(Ctxb, w_o, b_o, query, ln_g, ln_b, out);
}
